// GATLayer_15264313770199
// MI455X (gfx1250) — matrix operand order repaired and checked by reading, not compiled, not run on hardware
//
#include <hip/hip_runtime.h>
#include <hip/hip_bf16.h>
#include <math.h>

#define N_NODES 1024
#define IN_FEAT 256
#define N_HEADS 4
#define N_HIDDEN 32
#define G_COLS (N_HEADS * N_HIDDEN)   // 128
#define NEG_SLOPE 0.2f

typedef __attribute__((ext_vector_type(16))) _Float16 v16h;
typedef __attribute__((ext_vector_type(8)))  _Float16 v8h;
typedef __attribute__((ext_vector_type(8)))  float    v8f;

// Build the 16x32 f16 A-matrix fragment for one lane.
// Layout: lane half 0 -> K = {0..15}; half 1 -> K = {16..31} (matches contiguous B)
__device__ __forceinline__ v16h load_a_frag(const _Float16* __restrict__ row, int off) {
  v8h c0 = *reinterpret_cast<const v8h*>(row + off);
  v8h c1 = *reinterpret_cast<const v8h*>(row + off + 8);
  v16h r;
#pragma unroll
  for (int i = 0; i < 8; ++i) { r[i] = c0[i]; r[i + 8] = c1[i]; }
  return r;
}

// ---------------------------------------------------------------------------
// K0a: fp32 -> f16 convert of h
__global__ void cvt_h_kernel(const float* __restrict__ x, _Float16* __restrict__ y, int n) {
  int t = blockIdx.x * 256 + threadIdx.x;
  if (t < n) y[t] = (_Float16)x[t];
}

// K0b: W [256 x 128] -> WT f16 [128 x 256]  (B operand wants contiguous K)
__global__ void transpose_w_kernel(const float* __restrict__ W, _Float16* __restrict__ WT) {
  int t = blockIdx.x * 256 + threadIdx.x;  // 32768
  int n = t >> 8;      // 0..127 (output row = N)
  int k = t & 255;     // 0..255
  WT[t] = (_Float16)W[k * G_COLS + n];
}

// ---------------------------------------------------------------------------
// K1: g = h @ W via WMMA f16.  M=1024, N=128, K=256.
__global__ void gemm_g_kernel(const _Float16* __restrict__ h16,
                              const _Float16* __restrict__ wt16,
                              float* __restrict__ g,
                              _Float16* __restrict__ gT) {
  int wid  = blockIdx.x * 8 + (threadIdx.x >> 5);  // 0..511
  int lane = threadIdx.x & 31;
  int hl   = lane >> 4;
  int lr   = lane & 15;
  int mt   = wid >> 3;     // 64 M tiles
  int nt   = wid & 7;      // 8  N tiles
  int i0   = mt * 16;
  int n0   = nt * 16;

  const _Float16* arow = h16  + (i0 + lr) * IN_FEAT;
  const _Float16* brow = wt16 + (n0 + lr) * IN_FEAT;

  v8f acc = {};
#pragma unroll
  for (int kt = 0; kt < IN_FEAT / 32; ++kt) {
    int k0 = kt * 32;
    v16h a = load_a_frag(arow, k0 + hl * 16);
    v16h b = *reinterpret_cast<const v16h*>(brow + k0 + hl * 16);
    acc = __builtin_amdgcn_wmma_f32_16x16x32_f16(false, a, false, b,
                                                 (short)0, acc, false, false);
  }

  int head = nt >> 1;
  int fcol = (nt & 1) * 16 + lr;
#pragma unroll
  for (int r = 0; r < 8; ++r) {
    int i = i0 + r + 8 * hl;
    float v = acc[r];
    g[i * G_COLS + n0 + lr] = v;
    gT[(head * N_HIDDEN + fcol) * N_NODES + i] = (_Float16)v;
  }
}

// ---------------------------------------------------------------------------
// K1b: s[h][i] = sum_f g[i,h,f] * w[f]   (linear part of the leaky-relu split)
__global__ void s_kernel(const float* __restrict__ g, const float* __restrict__ aw,
                         float* __restrict__ s) {
  int t    = blockIdx.x * 256 + threadIdx.x;  // 4096
  int head = t >> 10;
  int i    = t & (N_NODES - 1);
  const float* gr = g + i * G_COLS + head * N_HIDDEN;
  float acc = 0.f;
#pragma unroll
  for (int f = 0; f < N_HIDDEN; ++f) acc = fmaf(gr[f], aw[f], acc);
  s[t] = acc;
}

// ---------------------------------------------------------------------------
// K2: scoring.  lrelu(x) = 0.6x + 0.4|x|  =>
//   e[i,j] = 0.6*(s_i + s_j) + 0.4 * sum_f |g_i + g_j|_f * w_f
// Block handles (h, 8 i-values, 256 j-values): g_j kept in VGPRs and reused
// across 8 i rows (8x less L2 traffic); inner loop = v_add + v_fmac(|src|).
__global__ void score_kernel(const float* __restrict__ g,
                             const float* __restrict__ s,
                             const int*   __restrict__ adj,
                             const float* __restrict__ aw,
                             float* __restrict__ e) {
  __shared__ float gi[8][N_HIDDEN];
  __shared__ float sish[8];
  __shared__ float w[N_HIDDEN];

  int b    = blockIdx.x;               // H * (N/8) * (N/256) = 2048
  int jb   = b & 3;
  int it   = (b >> 2) & 127;
  int head = b >> 9;
  int tid  = threadIdx.x;
  int j    = jb * 256 + tid;
  int i0   = it * 8;

  {
    int r = tid >> 5, f = tid & 31;    // 256 threads load 8x32 gi values
    gi[r][f] = g[(i0 + r) * G_COLS + head * N_HIDDEN + f];
    if (tid < N_HIDDEN) w[tid] = aw[tid];
    if (tid < 8)        sish[tid] = s[head * N_NODES + i0 + tid];
  }
  __syncthreads();

  float gj[N_HIDDEN];
  const float* gjp = g + j * G_COLS + head * N_HIDDEN;
#pragma unroll
  for (int f = 0; f < N_HIDDEN; ++f) gj[f] = gjp[f];
  float sj = s[head * N_NODES + j];

  float* erow = e + ((size_t)head * N_NODES + i0) * N_NODES + j;
#pragma unroll
  for (int r = 0; r < 8; ++r) {
    float acc = 0.f;
#pragma unroll
    for (int f = 0; f < N_HIDDEN; ++f)
      acc = fmaf(fabsf(gi[r][f] + gj[f]), w[f], acc);   // |src| is a free modifier
    float ev = fmaf(0.4f, acc, 0.6f * (sish[r] + sj));
    int aij = adj[(i0 + r) * N_NODES + j];
    erow[(size_t)r * N_NODES] = aij ? ev : -INFINITY;
  }
}

// ---------------------------------------------------------------------------
// K3: softmax over j per (h,i) row -> f16 attention weights.
// Row is staged into LDS with the CDNA5 async-copy path
// (global_load_async_to_lds_b128, ASYNCcnt), then reduced from LDS.
__global__ void softmax_kernel(const float* __restrict__ e, _Float16* __restrict__ a) {
  __shared__ float smrow[N_NODES];
  __shared__ float red[256];
  int row = blockIdx.x;           // h*N + i, 4096 rows
  int tid = threadIdx.x;
  const float* er = e + (size_t)row * N_NODES;

  // async copy: each lane moves 16B; block covers the full 4KB row
  unsigned     ldsoff = (unsigned)(size_t)(&smrow[tid * 4]);
  const float* gsrc   = er + tid * 4;
  asm volatile("global_load_async_to_lds_b128 %0, %1, off"
               :: "v"(ldsoff), "v"(gsrc) : "memory");
  asm volatile("s_wait_asynccnt 0x0" ::: "memory");
  __syncthreads();

  float v[4];
#pragma unroll
  for (int k = 0; k < 4; ++k) v[k] = smrow[tid + k * 256];

  float m = fmaxf(fmaxf(v[0], v[1]), fmaxf(v[2], v[3]));
  red[tid] = m;
  __syncthreads();
  for (int sN = 128; sN > 0; sN >>= 1) {
    if (tid < sN) red[tid] = fmaxf(red[tid], red[tid + sN]);
    __syncthreads();
  }
  m = red[0];
  __syncthreads();

  float ev[4], ssum = 0.f;
#pragma unroll
  for (int k = 0; k < 4; ++k) { ev[k] = __expf(v[k] - m); ssum += ev[k]; }
  red[tid] = ssum;
  __syncthreads();
  for (int sN = 128; sN > 0; sN >>= 1) {
    if (tid < sN) red[tid] += red[tid + sN];
    __syncthreads();
  }
  float inv = 1.0f / red[0];

  _Float16* ar = a + (size_t)row * N_NODES;
#pragma unroll
  for (int k = 0; k < 4; ++k) ar[tid + k * 256] = (_Float16)(ev[k] * inv);
}

// ---------------------------------------------------------------------------
// K4: per-head aggregation  attn[h] = A_h(1024x1024) @ G_h(1024x32) via WMMA.
__global__ void agg_kernel(const _Float16* __restrict__ a16,
                           const _Float16* __restrict__ gT,
                           float* __restrict__ attn) {
  int wid  = blockIdx.x * 8 + (threadIdx.x >> 5);  // 0..511
  int lane = threadIdx.x & 31;
  int hl   = lane >> 4;
  int lr   = lane & 15;
  int head = wid >> 7;
  int rem  = wid & 127;
  int mt   = rem >> 1;
  int nt   = rem & 1;
  int i0   = mt * 16;
  int f0   = nt * 16;

  const _Float16* arow = a16 + ((size_t)(head * N_NODES + i0 + lr)) * N_NODES;
  const _Float16* brow = gT  + (head * N_HIDDEN + f0 + lr) * N_NODES;

  v8f acc = {};
#pragma unroll 4
  for (int kt = 0; kt < N_NODES / 32; ++kt) {
    int k0 = kt * 32;
    v16h a = load_a_frag(arow, k0 + hl * 16);
    v16h b = *reinterpret_cast<const v16h*>(brow + k0 + hl * 16);
    acc = __builtin_amdgcn_wmma_f32_16x16x32_f16(false, a, false, b,
                                                 (short)0, acc, false, false);
  }

#pragma unroll
  for (int r = 0; r < 8; ++r) {
    int i = i0 + r + 8 * hl;
    attn[((size_t)head * N_NODES + i) * N_HIDDEN + f0 + lr] = acc[r];
  }
}

// ---------------------------------------------------------------------------
// K5: mean over heads (deterministic, fixed order).
__global__ void mean_kernel(const float* __restrict__ attn, float* __restrict__ out) {
  int t = blockIdx.x * 256 + threadIdx.x;  // 32768
  int i = t >> 5;
  int f = t & 31;
  float s = 0.f;
#pragma unroll
  for (int hh = 0; hh < N_HEADS; ++hh)
    s += attn[((size_t)hh * N_NODES + i) * N_HIDDEN + f];
  out[t] = 0.25f * s;
}

// ---------------------------------------------------------------------------
extern "C" void kernel_launch(void* const* d_in, const int* in_sizes, int n_in,
                              void* d_out, int out_size, void* d_ws, size_t ws_size,
                              hipStream_t stream) {
  const float* h   = (const float*)d_in[0];
  const int*   adj = (const int*)  d_in[1];
  const float* W   = (const float*)d_in[2];
  const float* aw  = (const float*)d_in[3];
  float* out = (float*)d_out;

  char* ws = (char*)d_ws;
  size_t off = 0;
  auto alloc = [&](size_t bytes) -> void* {
    void* p = ws + off;
    off += (bytes + 255) & ~(size_t)255;
    return p;
  };

  _Float16* h16  = (_Float16*)alloc((size_t)N_NODES * IN_FEAT * 2);             // 512 KB
  _Float16* wt16 = (_Float16*)alloc((size_t)G_COLS * IN_FEAT * 2);              // 64 KB
  float*    g    = (float*)   alloc((size_t)N_NODES * G_COLS * 4);              // 512 KB
  _Float16* gT   = (_Float16*)alloc((size_t)N_HEADS * N_HIDDEN * N_NODES * 2);  // 256 KB
  float*    sbuf = (float*)   alloc((size_t)N_HEADS * N_NODES * 4);             // 16 KB
  float*    e    = (float*)   alloc((size_t)N_HEADS * N_NODES * N_NODES * 4);   // 16 MB
  _Float16* a16  = (_Float16*)alloc((size_t)N_HEADS * N_NODES * N_NODES * 2);   // 8 MB
  float*    attn = (float*)   alloc((size_t)N_HEADS * N_NODES * N_HIDDEN * 4);  // 512 KB

  cvt_h_kernel      <<<(N_NODES * IN_FEAT + 255) / 256, 256, 0, stream>>>(h, h16, N_NODES * IN_FEAT);
  transpose_w_kernel<<<(G_COLS * IN_FEAT) / 256,        256, 0, stream>>>(W, wt16);
  gemm_g_kernel     <<<64,                              256, 0, stream>>>(h16, wt16, g, gT);
  s_kernel          <<<(N_HEADS * N_NODES) / 256,       256, 0, stream>>>(g, aw, sbuf);
  score_kernel      <<<N_HEADS * (N_NODES / 8) * (N_NODES / 256), 256, 0, stream>>>(g, sbuf, adj, aw, e);
  softmax_kernel    <<<N_HEADS * N_NODES,               256, 0, stream>>>(e, a16);
  agg_kernel        <<<64,                              256, 0, stream>>>(a16, gT, attn);
  mean_kernel       <<<(N_NODES * N_HIDDEN) / 256,      256, 0, stream>>>(attn, out);
}
